// ActivationGatedGCNNet_90993177133098
// MI455X (gfx1250) — matrix operand order repaired and checked by reading, not compiled, not run on hardware
//
#include <hip/hip_runtime.h>

#define NNODES   50000
#define NEDGES   800000
#define NGRAPHS  256
#define HID      64
#define NLAYERS  4
#define BN_EPS   1e-5f
#define GATE_EPS 1e-6f

typedef __attribute__((ext_vector_type(16))) _Float16 v16h;
typedef __attribute__((ext_vector_type(8)))  float    v8f;

__device__ __forceinline__ float sigmoidf_(float x) { return 1.0f / (1.0f + __expf(-x)); }

// ---------------- generic helpers ----------------
__global__ void k_zero(float* __restrict__ p, int n) {
  int i = blockIdx.x * blockDim.x + threadIdx.x;
  if (i < n) p[i] = 0.0f;
}

__global__ void k_embed(const int* __restrict__ idx, const float* __restrict__ emb,
                        float* __restrict__ out, int n) {
  int i = blockIdx.x * blockDim.x + threadIdx.x;
  if (i < n) out[i] = emb[idx[i >> 6] * HID + (i & 63)];
}

__global__ void k_deg(const int* __restrict__ dst, float* __restrict__ deg) {
  int e = blockIdx.x * blockDim.x + threadIdx.x;
  if (e < NEDGES) atomicAdd(&deg[dst[e]], 1.0f);
}

__global__ void k_norm(const float* __restrict__ deg, float* __restrict__ nrm) {
  int i = blockIdx.x * blockDim.x + threadIdx.x;
  if (i < NNODES) nrm[i] = rsqrtf(fmaxf(deg[i], 1.0f));
}

// ---------------- layer pass 1: edge gate + scatter + edge BN stats ----------------
__global__ void k_edge_gate(const int* __restrict__ src, const int* __restrict__ dst,
                            const float* __restrict__ hf, const float* __restrict__ ef,
                            const float* __restrict__ nrm,
                            float* __restrict__ msg, float* __restrict__ den,
                            float* __restrict__ stats) {
  __shared__ float s_sum[HID];
  __shared__ float s_sq[HID];
  const int t = threadIdx.x;
  if (t < HID) { s_sum[t] = 0.0f; s_sq[t] = 0.0f; }
  __syncthreads();
  float sx=0.f,sy=0.f,sz=0.f,sw=0.f, qx=0.f,qy=0.f,qz=0.f,qw=0.f;
  const int stride = gridDim.x * blockDim.x;
  for (int slot = blockIdx.x * blockDim.x + t; slot < NEDGES * 16; slot += stride) {
    const int e = slot >> 4, l = slot & 15;
    const int s = src[e], d = dst[e];
    const float4 a = *(const float4*)(hf + s * HID + l * 4);
    const float4 b = *(const float4*)(hf + d * HID + l * 4);
    const float4 c = *(const float4*)(ef + e * HID + l * 4);
    const float ex = a.x + b.x + c.x, ey = a.y + b.y + c.y;
    const float ez = a.z + b.z + c.z, ew = a.w + b.w + c.w;
    const float gx = sigmoidf_(ex), gy = sigmoidf_(ey);
    const float gz = sigmoidf_(ez), gw = sigmoidf_(ew);
    const float ns = nrm[s];
    float* dp = den + d * HID + l * 4;
    atomicAdd(dp + 0, gx); atomicAdd(dp + 1, gy);
    atomicAdd(dp + 2, gz); atomicAdd(dp + 3, gw);
    float* mp = msg + d * HID + l * 4;
    atomicAdd(mp + 0, gx * a.x * ns); atomicAdd(mp + 1, gy * a.y * ns);
    atomicAdd(mp + 2, gz * a.z * ns); atomicAdd(mp + 3, gw * a.w * ns);
    sx += ex; sy += ey; sz += ez; sw += ew;
    qx += ex * ex; qy += ey * ey; qz += ez * ez; qw += ew * ew;
  }
  const int fb = (t & 15) * 4;
  atomicAdd(&s_sum[fb + 0], sx); atomicAdd(&s_sum[fb + 1], sy);
  atomicAdd(&s_sum[fb + 2], sz); atomicAdd(&s_sum[fb + 3], sw);
  atomicAdd(&s_sq[fb + 0], qx);  atomicAdd(&s_sq[fb + 1], qy);
  atomicAdd(&s_sq[fb + 2], qz);  atomicAdd(&s_sq[fb + 3], qw);
  __syncthreads();
  if (t < HID) { atomicAdd(&stats[128 + t], s_sum[t]); atomicAdd(&stats[192 + t], s_sq[t]); }
}

// ---------------- layer pass 2: node update + node BN stats ----------------
__global__ void k_node_update(const float* __restrict__ hf, const float* __restrict__ nrm,
                              const float* __restrict__ msg, const float* __restrict__ den,
                              float* __restrict__ h_tmp, float* __restrict__ stats) {
  __shared__ float s_sum[HID];
  __shared__ float s_sq[HID];
  const int t = threadIdx.x;
  if (t < HID) { s_sum[t] = 0.0f; s_sq[t] = 0.0f; }
  __syncthreads();
  float sx=0.f,sy=0.f,sz=0.f,sw=0.f, qx=0.f,qy=0.f,qz=0.f,qw=0.f;
  const int stride = gridDim.x * blockDim.x;
  for (int slot = blockIdx.x * blockDim.x + t; slot < NNODES * 16; slot += stride) {
    const int n = slot >> 4, l = slot & 15;
    const float r = nrm[n];
    const float4 hv = *(const float4*)(hf + n * HID + l * 4);
    const float4 m  = *(const float4*)(msg + n * HID + l * 4);
    const float4 dn = *(const float4*)(den + n * HID + l * 4);
    const float ox = (hv.x * r + m.x / (dn.x + GATE_EPS)) * r;
    const float oy = (hv.y * r + m.y / (dn.y + GATE_EPS)) * r;
    const float oz = (hv.z * r + m.z / (dn.z + GATE_EPS)) * r;
    const float ow = (hv.w * r + m.w / (dn.w + GATE_EPS)) * r;
    float4 o; o.x = ox; o.y = oy; o.z = oz; o.w = ow;
    *(float4*)(h_tmp + n * HID + l * 4) = o;
    sx += ox; sy += oy; sz += oz; sw += ow;
    qx += ox * ox; qy += oy * oy; qz += oz * oz; qw += ow * ow;
  }
  const int fb = (t & 15) * 4;
  atomicAdd(&s_sum[fb + 0], sx); atomicAdd(&s_sum[fb + 1], sy);
  atomicAdd(&s_sum[fb + 2], sz); atomicAdd(&s_sum[fb + 3], sw);
  atomicAdd(&s_sq[fb + 0], qx);  atomicAdd(&s_sq[fb + 1], qy);
  atomicAdd(&s_sq[fb + 2], qz);  atomicAdd(&s_sq[fb + 3], qw);
  __syncthreads();
  if (t < HID) { atomicAdd(&stats[t], s_sum[t]); atomicAdd(&stats[64 + t], s_sq[t]); }
}

// ---------------- fold BN into affine coefficients ----------------
__global__ void k_bn_coeff(const float* __restrict__ stats,
                           const float* __restrict__ gh, const float* __restrict__ bh,
                           const float* __restrict__ ge, const float* __restrict__ be,
                           int layer, float* __restrict__ coeff) {
  const int t = threadIdx.x;  // 128 threads
  if (t < HID) {
    const float mu  = stats[t] * (1.0f / (float)NNODES);
    const float var = fmaxf(stats[64 + t] * (1.0f / (float)NNODES) - mu * mu, 0.0f);
    const float a   = gh[layer * HID + t] * rsqrtf(var + BN_EPS);
    coeff[t] = a; coeff[64 + t] = bh[layer * HID + t] - a * mu;
  } else {
    const int f = t - HID;
    const float mu  = stats[128 + f] * (1.0f / (float)NEDGES);
    const float var = fmaxf(stats[192 + f] * (1.0f / (float)NEDGES) - mu * mu, 0.0f);
    const float a   = ge[layer * HID + f] * rsqrtf(var + BN_EPS);
    coeff[128 + f] = a; coeff[192 + f] = be[layer * HID + f] - a * mu;
  }
}

// ---------------- apply passes (e_new recomputed from L2-resident hf) ----------------
__global__ void k_edge_apply(const int* __restrict__ src, const int* __restrict__ dst,
                             const float* __restrict__ hf, float* __restrict__ ef,
                             const float* __restrict__ coeff) {
  const int slot = blockIdx.x * blockDim.x + threadIdx.x;
  if (slot >= NEDGES * 16) return;
  const int e = slot >> 4, l = slot & 15;
  const int s = src[e], d = dst[e];
  const float4 a = *(const float4*)(hf + s * HID + l * 4);
  const float4 b = *(const float4*)(hf + d * HID + l * 4);
  float* ep = ef + e * HID + l * 4;
  float4 c = *(const float4*)ep;
  const float4 ae = *(const float4*)(coeff + 128 + l * 4);
  const float4 bv = *(const float4*)(coeff + 192 + l * 4);
  c.x += fmaxf(ae.x * (a.x + b.x + c.x) + bv.x, 0.0f);
  c.y += fmaxf(ae.y * (a.y + b.y + c.y) + bv.y, 0.0f);
  c.z += fmaxf(ae.z * (a.z + b.z + c.z) + bv.z, 0.0f);
  c.w += fmaxf(ae.w * (a.w + b.w + c.w) + bv.w, 0.0f);
  *(float4*)ep = c;
}

__global__ void k_node_apply(const float* __restrict__ h_tmp, const float* __restrict__ coeff,
                             float* __restrict__ hf) {
  const int slot = blockIdx.x * blockDim.x + threadIdx.x;
  if (slot >= NNODES * 16) return;
  const int n = slot >> 4, l = slot & 15;
  const float4 ht = *(const float4*)(h_tmp + n * HID + l * 4);
  const float4 ah = *(const float4*)(coeff + l * 4);
  const float4 bh = *(const float4*)(coeff + 64 + l * 4);
  float* hp = hf + n * HID + l * 4;
  float4 hv = *(const float4*)hp;
  hv.x += fmaxf(ah.x * ht.x + bh.x, 0.0f);
  hv.y += fmaxf(ah.y * ht.y + bh.y, 0.0f);
  hv.z += fmaxf(ah.z * ht.z + bh.z, 0.0f);
  hv.w += fmaxf(ah.w * ht.w + bh.w, 0.0f);
  *(float4*)hp = hv;
}

// ---------------- graph mean pooling ----------------
__global__ void k_pool(const float* __restrict__ hf, const int* __restrict__ gid,
                       float* __restrict__ hg, float* __restrict__ counts) {
  const int slot = blockIdx.x * blockDim.x + threadIdx.x;
  if (slot >= NNODES * 16) return;
  const int n = slot >> 4, l = slot & 15;
  const int g = gid[n];
  const float4 v = *(const float4*)(hf + n * HID + l * 4);
  float* hp = hg + g * HID + l * 4;
  atomicAdd(hp + 0, v.x); atomicAdd(hp + 1, v.y);
  atomicAdd(hp + 2, v.z); atomicAdd(hp + 3, v.w);
  if (l == 0) atomicAdd(&counts[g], 1.0f);
}

__global__ void k_pool_div(float* __restrict__ hg, const float* __restrict__ counts) {
  const int i = blockIdx.x * blockDim.x + threadIdx.x;
  if (i < NGRAPHS * HID) hg[i] /= fmaxf(counts[i >> 6], 1.0f);
}

// ---------------- MLP head via WMMA f16 (fragment layouts used here; A/B in contiguous k order) ----------------
// A (16x32 f16): elem i of lane -> K = i + 16*(lane/16),              M = lane%16
// B (32x16 f16): elem i of lane -> K = i + 16*(lane/16),              N = lane%16
// C/D (16x16 f32): vgpr r of lane -> M = r + 8*(lane/16),             N = lane%16
__global__ void k_mlp1(const float* __restrict__ hg, const float* __restrict__ W1,
                       const float* __restrict__ b1, _Float16* __restrict__ x1) {
  const int lane = threadIdx.x & 31, wid = threadIdx.x >> 5;   // 32 waves
  const int mTile = wid >> 1, nTile = wid & 1;
  const int M = lane & 15, half = lane >> 4, Ncol = lane & 15;
  v16h a0, a1, w0, w1;
#pragma unroll
  for (int i = 0; i < 16; ++i) {
    const int kA = i + 16 * half;
    a0[i] = (_Float16)hg[(mTile * 16 + M) * HID + kA];
    a1[i] = (_Float16)hg[(mTile * 16 + M) * HID + 32 + kA];
    const int kB = i + 16 * half;
    w0[i] = (_Float16)W1[kB * 32 + nTile * 16 + Ncol];
    w1[i] = (_Float16)W1[(kB + 32) * 32 + nTile * 16 + Ncol];
  }
  v8f c = {};
  c = __builtin_amdgcn_wmma_f32_16x16x32_f16(false, a0, false, w0, (short)0, c, false, false);
  c = __builtin_amdgcn_wmma_f32_16x16x32_f16(false, a1, false, w1, (short)0, c, false, false);
  const float bias = b1[nTile * 16 + Ncol];
#pragma unroll
  for (int r = 0; r < 8; ++r) {
    const int row = mTile * 16 + r + 8 * half;
    x1[row * 32 + nTile * 16 + Ncol] = (_Float16)fmaxf(c[r] + bias, 0.0f);
  }
}

__global__ void k_mlp2(const _Float16* __restrict__ x1, const float* __restrict__ W2,
                       const float* __restrict__ b2, _Float16* __restrict__ x2) {
  const int lane = threadIdx.x & 31, mTile = threadIdx.x >> 5;  // 16 waves
  const int M = lane & 15, half = lane >> 4, Ncol = lane & 15;
  v16h a, w;
#pragma unroll
  for (int i = 0; i < 16; ++i) {
    const int kA = i + 16 * half;
    a[i] = x1[(mTile * 16 + M) * 32 + kA];
    const int kB = i + 16 * half;
    w[i] = (_Float16)W2[kB * 16 + Ncol];
  }
  v8f c = {};
  c = __builtin_amdgcn_wmma_f32_16x16x32_f16(false, a, false, w, (short)0, c, false, false);
  const float bias = b2[Ncol];
#pragma unroll
  for (int r = 0; r < 8; ++r) {
    const int row = mTile * 16 + r + 8 * half;
    x2[row * 16 + Ncol] = (_Float16)fmaxf(c[r] + bias, 0.0f);
  }
}

__global__ void k_mlp3(const _Float16* __restrict__ x2, const float* __restrict__ W3,
                       const float* __restrict__ b3, float* __restrict__ out) {
  const int g = threadIdx.x;  // 256 threads, 1 block
  float acc = b3[0];
#pragma unroll
  for (int k = 0; k < 16; ++k) acc += (float)x2[g * 16 + k] * W3[k];
  out[g] = acc;
}

// ---------------- host-side orchestration ----------------
extern "C" void kernel_launch(void* const* d_in, const int* in_sizes, int n_in,
                              void* d_out, int out_size, void* d_ws, size_t ws_size,
                              hipStream_t stream) {
  (void)in_sizes; (void)n_in; (void)out_size; (void)ws_size;
  const int*   h        = (const int*)d_in[0];
  const int*   e        = (const int*)d_in[1];
  const int*   src      = (const int*)d_in[2];
  const int*   dst      = (const int*)d_in[3];
  const int*   gid      = (const int*)d_in[4];
  const float* node_emb = (const float*)d_in[5];
  const float* edge_emb = (const float*)d_in[6];
  const float* gh       = (const float*)d_in[7];
  const float* bh       = (const float*)d_in[8];
  const float* ge       = (const float*)d_in[9];
  const float* be       = (const float*)d_in[10];
  const float* W1       = (const float*)d_in[11];
  const float* b1       = (const float*)d_in[12];
  const float* W2       = (const float*)d_in[13];
  const float* b2       = (const float*)d_in[14];
  const float* W3       = (const float*)d_in[15];
  const float* b3       = (const float*)d_in[16];
  float* out = (float*)d_out;

  float* ws = (float*)d_ws;
  size_t off = 0;
  float* hf     = ws + off; off += (size_t)NNODES * HID;   // 12.8 MB  (L2-resident)
  float* ef     = ws + off; off += (size_t)NEDGES * HID;   // 204.8 MB (streams)
  float* h_tmp  = ws + off; off += (size_t)NNODES * HID;
  float* msg    = ws + off; off += (size_t)NNODES * HID;   // msg, den contiguous
  float* den    = ws + off; off += (size_t)NNODES * HID;
  float* nrm    = ws + off; off += NNODES;
  float* stats  = ws + off; off += 256;   // [h_sum|h_sq|e_sum|e_sq] x64
  float* coeff  = ws + off; off += 256;   // [a_h|b_h|a_e|b_e] x64
  float* hg     = ws + off; off += NGRAPHS * HID;          // hg, counts contiguous
  float* counts = ws + off; off += NGRAPHS;
  _Float16* x1  = (_Float16*)(ws + off); off += NGRAPHS * 32 / 2;
  _Float16* x2  = (_Float16*)(ws + off); off += NGRAPHS * 16 / 2;

  const int T = 256;
  // Encoders
  k_embed<<<(NNODES * HID + T - 1) / T, T, 0, stream>>>(h, node_emb, hf, NNODES * HID);
  k_embed<<<(NEDGES * HID + T - 1) / T, T, 0, stream>>>(e, edge_emb, ef, NEDGES * HID);
  // Degrees -> norm (msg buffer reused as degree accumulator)
  k_zero<<<(NNODES + T - 1) / T, T, 0, stream>>>(msg, NNODES);
  k_deg<<<(NEDGES + T - 1) / T, T, 0, stream>>>(dst, msg);
  k_norm<<<(NNODES + T - 1) / T, T, 0, stream>>>(msg, nrm);

  for (int l = 0; l < NLAYERS; ++l) {
    k_zero<<<(2 * NNODES * HID + T - 1) / T, T, 0, stream>>>(msg, 2 * NNODES * HID);
    k_zero<<<1, 256, 0, stream>>>(stats, 256);
    k_edge_gate<<<4096, T, 0, stream>>>(src, dst, hf, ef, nrm, msg, den, stats);
    k_node_update<<<2048, T, 0, stream>>>(hf, nrm, msg, den, h_tmp, stats);
    k_bn_coeff<<<1, 128, 0, stream>>>(stats, gh, bh, ge, be, l, coeff);
    k_edge_apply<<<(NEDGES * 16 + T - 1) / T, T, 0, stream>>>(src, dst, hf, ef, coeff);
    k_node_apply<<<(NNODES * 16 + T - 1) / T, T, 0, stream>>>(h_tmp, coeff, hf);
  }

  // Per-graph mean pooling
  k_zero<<<(NGRAPHS * HID + NGRAPHS + T - 1) / T, T, 0, stream>>>(hg, NGRAPHS * HID + NGRAPHS);
  k_pool<<<(NNODES * 16 + T - 1) / T, T, 0, stream>>>(hf, gid, hg, counts);
  k_pool_div<<<(NGRAPHS * HID + T - 1) / T, T, 0, stream>>>(hg, counts);

  // MLP head (WMMA)
  k_mlp1<<<1, 1024, 0, stream>>>(hg, W1, b1, x1);
  k_mlp2<<<1, 512, 0, stream>>>(x1, W2, b2, x2);
  k_mlp3<<<1, 256, 0, stream>>>(x2, W3, b3, out);
}
